// BiDirectionalBlock_76441827934440
// MI455X (gfx1250) — compile-verified, matrix operand order checked by reading, not run on hardware
//
#include <hip/hip_runtime.h>

// ---------------------------------------------------------------------------
// Bidirectional Mamba block for gfx1250 (MI455X).
// GEMMs: bf16 WMMA 16x16x32 (f32 accum), tiles staged with
//        global_load_async_to_lds_b128 (ASYNCcnt), double-buffered LDS.
// Scan:  wave-per-(b,d) recurrence, lane-per-2-states, shfl butterfly.
// ---------------------------------------------------------------------------

#define D_MODEL   1024
#define D_INNER   2048
#define D_STATE   64
#define DT_RANK   64
#define SEQ_L     2048
#define BATCH     4
#define ROWS      (BATCH * SEQ_L)          // 8192
#define DBL_N     (DT_RANK + 2 * D_STATE)  // 192

#define BM 128
#define BN 128
#define KC 32
#define LDT 40                              // padded LDS row stride (ushorts)

typedef __attribute__((ext_vector_type(16))) __bf16 v16bf;
typedef __attribute__((ext_vector_type(8)))  float  v8f;
typedef __attribute__((ext_vector_type(8)))  unsigned short ushort8;

union Frag {
    v16bf   v;
    ushort8 h[2];
};

__device__ __forceinline__ unsigned short f2bf(float f) {
    union { float f; unsigned u; } c;
    c.f = f;
    unsigned u = c.u;
    u += 0x7fffu + ((u >> 16) & 1u);        // round-to-nearest-even
    return (unsigned short)(u >> 16);
}

// ---------------------------------------------------------------------------
// GEMM: C(8192 x N) = A(bf16, row stride lda) * W(bf16, NxK row-major)^T
// W rows are padded to a multiple of BN (zero-filled) so tile staging is
// unguarded. EPI: 0 = f32 store, 1 = +bias softplus f32, 2 = +bias +resid f32,
// 3 = bf16 store.
// ---------------------------------------------------------------------------
template <int EPI>
__global__ __launch_bounds__(256) void gemm_async_wmma(
    const unsigned short* __restrict__ A,
    const unsigned short* __restrict__ W,
    const float* __restrict__ bias,
    const float* __restrict__ resid,
    void* __restrict__ Cout,
    int N, int K, int lda, int ldc)
{
    __shared__ __attribute__((aligned(16))) unsigned short As[2][BM * LDT];
    __shared__ __attribute__((aligned(16))) unsigned short Bs[2][BN * LDT];

    const int tid  = threadIdx.x;
    const int lane = tid & 31;
    const int wave = tid >> 5;        // 0..7
    const int wm   = wave >> 2;       // 0..1 along M
    const int wn   = wave & 3;        // 0..3 along N
    const int blockM = blockIdx.y * BM;
    const int blockN = blockIdx.x * BN;

    // Low 32 bits of a generic LDS pointer = wave-relative LDS byte address.
    const unsigned asBase[2] = { (unsigned)(size_t)&As[0][0],
                                 (unsigned)(size_t)&As[1][0] };
    const unsigned bsBase[2] = { (unsigned)(size_t)&Bs[0][0],
                                 (unsigned)(size_t)&Bs[1][0] };
    const unsigned long long abase = (unsigned long long)A;
    const unsigned long long wbase = (unsigned long long)W;

    // Stage one KC-chunk of A and W tiles into LDS buffer `bufi` via the
    // async global->LDS path. 512 16B segments per tile, 256 threads -> each
    // thread issues 2 A + 2 W copies (tracked on ASYNCcnt).
    auto issue = [&](int bufi, int k0) {
#pragma unroll
        for (int i = 0; i < 2; ++i) {
            int seg = tid + i * 256;            // 0..511
            int row = seg >> 2, s = seg & 3;    // 4 x 16B segments per row
            unsigned lds  = asBase[bufi] + (unsigned)(row * (LDT * 2) + s * 16);
            unsigned voff = (unsigned)(((size_t)(blockM + row) * lda + k0 + s * 8) * 2);
            asm volatile("global_load_async_to_lds_b128 %0, %1, %2"
                         :: "v"(lds), "v"(voff), "s"(abase) : "memory");
        }
#pragma unroll
        for (int i = 0; i < 2; ++i) {
            int seg = tid + i * 256;
            int row = seg >> 2, s = seg & 3;
            unsigned lds  = bsBase[bufi] + (unsigned)(row * (LDT * 2) + s * 16);
            unsigned voff = (unsigned)(((size_t)(blockN + row) * K + k0 + s * 8) * 2);
            asm volatile("global_load_async_to_lds_b128 %0, %1, %2"
                         :: "v"(lds), "v"(voff), "s"(wbase) : "memory");
        }
    };

    v8f acc[4][2];
#pragma unroll
    for (int i = 0; i < 4; ++i)
#pragma unroll
        for (int j = 0; j < 2; ++j)
            acc[i][j] = (v8f){0.f, 0.f, 0.f, 0.f, 0.f, 0.f, 0.f, 0.f};

    const int khalf = lane >> 4;
    const int r16   = lane & 15;

    issue(0, 0);
    int buf = 0;
    for (int k0 = 0; k0 < K; k0 += KC) {
        asm volatile("s_wait_asynccnt 0x0" ::: "memory");   // my copies landed
        __syncthreads();                                    // everyone's landed
        if (k0 + KC < K) issue(buf ^ 1, k0 + KC);           // overlap next chunk

        const unsigned short* as = As[buf];
        const unsigned short* bs = Bs[buf];
        Frag af[4], bf[2];
#pragma unroll
        for (int fm = 0; fm < 4; ++fm) {
            int row = wm * 64 + fm * 16 + r16;
            af[fm].h[0] = *(const ushort8*)&as[row * LDT + khalf * 8];
            af[fm].h[1] = *(const ushort8*)&as[row * LDT + 16 + khalf * 8];
        }
#pragma unroll
        for (int fn = 0; fn < 2; ++fn) {
            int row = wn * 32 + fn * 16 + r16;
            bf[fn].h[0] = *(const ushort8*)&bs[row * LDT + khalf * 8];
            bf[fn].h[1] = *(const ushort8*)&bs[row * LDT + 16 + khalf * 8];
        }
#pragma unroll
        for (int fm = 0; fm < 4; ++fm)
#pragma unroll
            for (int fn = 0; fn < 2; ++fn)
                acc[fm][fn] = __builtin_amdgcn_wmma_f32_16x16x32_bf16(
                    false, af[fm].v, false, bf[fn].v,
                    (short)0, acc[fm][fn], false, false);
        buf ^= 1;
    }

    // D layout: lane<16 -> M=r, lane>=16 -> M=r+8 ; N = lane&15
    const int mo = (lane >> 4) * 8;
    float* Cf = (float*)Cout;
    unsigned short* Cb = (unsigned short*)Cout;
#pragma unroll
    for (int fm = 0; fm < 4; ++fm)
#pragma unroll
        for (int fn = 0; fn < 2; ++fn) {
            int col = blockN + wn * 32 + fn * 16 + r16;
            if (col >= N) continue;
            int rowBase = blockM + wm * 64 + fm * 16 + mo;
#pragma unroll
            for (int r = 0; r < 8; ++r) {
                float v = acc[fm][fn][r];
                int row = rowBase + r;
                if (EPI == 1) {
                    v += bias[col];
                    v = (v > 20.f) ? v : __logf(1.f + __expf(v)); // softplus
                } else if (EPI == 2) {
                    v += bias[col] + resid[(size_t)row * ldc + col];
                }
                if (EPI == 3) Cb[(size_t)row * ldc + col] = f2bf(v);
                else          Cf[(size_t)row * ldc + col] = v;
            }
        }
}

// ---------------------------------------------------------------------------
// Pack helpers (f32 -> bf16)
// ---------------------------------------------------------------------------
__global__ __launch_bounds__(256) void pack_bf16_kernel(
    const float* __restrict__ src, unsigned short* __restrict__ dst)
{
    int idx = blockIdx.x * 256 + threadIdx.x;
    dst[idx] = f2bf(src[idx]);
}

// Pack weight (NxK) into Npad x K (grid covers Npad*K), zero rows >= N.
__global__ __launch_bounds__(256) void pack_weight_kernel(
    const float* __restrict__ src, unsigned short* __restrict__ dst,
    int N, int K)
{
    int idx = blockIdx.x * 256 + threadIdx.x;
    int row = idx / K, c = idx - row * K;
    dst[idx] = (row < N) ? f2bf(src[(size_t)row * K + c]) : (unsigned short)0;
}

// Extract dt slice: dbl(ROWS x 192)[:, 0:64] -> dtbf(ROWS x 64) bf16.
__global__ __launch_bounds__(256) void pack_dt_kernel(
    const float* __restrict__ dbl, unsigned short* __restrict__ dtbf)
{
    int idx = blockIdx.x * 256 + threadIdx.x;   // ROWS*64
    int row = idx >> 6, c = idx & 63;
    dtbf[idx] = f2bf(dbl[(size_t)row * DBL_N + c]);
}

// ---------------------------------------------------------------------------
// Depthwise causal conv (K=4) + SiLU; writes f32 (for scan) and bf16 (for the
// x_proj GEMM). dir=0: taps l-3..l w[0..3]; dir=1: taps l..l+3 w[3..0].
// ---------------------------------------------------------------------------
__global__ __launch_bounds__(256) void conv_silu_kernel(
    const float* __restrict__ xz,    // (ROWS, 4096), xi = cols [0,2048)
    const float* __restrict__ cw,    // (D_INNER, 4)
    const float* __restrict__ cb,    // (D_INNER)
    float* __restrict__ xc,          // (ROWS, D_INNER) f32
    unsigned short* __restrict__ xcbf,
    int dir)
{
    int idx = blockIdx.x * 256 + threadIdx.x;   // ROWS * D_INNER
    int d = idx & (D_INNER - 1);
    int l = (idx >> 11) & (SEQ_L - 1);
    int b = idx >> 22;
    const float* w = cw + d * 4;
    float acc = cb[d];
    size_t base = (size_t)b * SEQ_L * (2 * D_INNER) + d;
    if (dir == 0) {
#pragma unroll
        for (int k = 0; k < 4; ++k) {
            int ll = l - 3 + k;
            if (ll >= 0) acc += xz[base + (size_t)ll * (2 * D_INNER)] * w[k];
        }
    } else {
#pragma unroll
        for (int o = 0; o < 4; ++o) {
            int ll = l + o;
            if (ll < SEQ_L) acc += xz[base + (size_t)ll * (2 * D_INNER)] * w[3 - o];
        }
    }
    float sig = 1.f / (1.f + __expf(-acc));
    float v = acc * sig;
    size_t o = (size_t)(b * SEQ_L + l) * D_INNER + d;
    xc[o]   = v;
    xcbf[o] = f2bf(v);
}

// ---------------------------------------------------------------------------
// Selective scan: one wave per (b,d); lane handles states {lane, lane+32}.
// Output (gated, +x*D) written as bf16 for the out_proj GEMM.
// ---------------------------------------------------------------------------
__global__ __launch_bounds__(256) void scan_kernel(
    const float* __restrict__ delta, // (ROWS, D_INNER)
    const float* __restrict__ dbl,   // (ROWS, 192): dt|B|C
    const float* __restrict__ xz,    // (ROWS, 4096): z at col offset 2048
    const float* __restrict__ xc,    // (ROWS, D_INNER)
    const float* __restrict__ A_log, // (D_INNER, D_STATE)
    const float* __restrict__ Dp,    // (D_INNER)
    unsigned short* __restrict__ ybf,// (ROWS, D_INNER) bf16
    int dir)
{
    int w    = blockIdx.x * 8 + (threadIdx.x >> 5);
    int lane = threadIdx.x & 31;
    int b = w >> 11;
    int d = w & (D_INNER - 1);
    int s0 = lane, s1 = lane + 32;

    float a0 = -__expf(A_log[d * D_STATE + s0]);
    float a1 = -__expf(A_log[d * D_STATE + s1]);
    float Dv = Dp[d];
    float h0 = 0.f, h1 = 0.f;

    int l   = dir ? (SEQ_L - 1) : 0;
    int stp = dir ? -1 : 1;
    for (int t = 0; t < SEQ_L; ++t, l += stp) {
        size_t row = (size_t)b * SEQ_L + l;
        float dt = delta[row * D_INNER + d];
        float xv = xc[row * D_INNER + d];
        const float* bc = dbl + row * DBL_N;
        float B0 = bc[DT_RANK + s0], B1 = bc[DT_RANK + s1];
        float C0 = bc[DT_RANK + D_STATE + s0], C1 = bc[DT_RANK + D_STATE + s1];
        float dx = dt * xv;
        h0 = __expf(dt * a0) * h0 + dx * B0;
        h1 = __expf(dt * a1) * h1 + dx * B1;
        float y = h0 * C0 + h1 * C1;
        y += __shfl_xor(y, 16);
        y += __shfl_xor(y, 8);
        y += __shfl_xor(y, 4);
        y += __shfl_xor(y, 2);
        y += __shfl_xor(y, 1);
        if (lane == 0) {
            float z = xz[row * (2 * D_INNER) + D_INNER + d];
            float sig = 1.f / (1.f + __expf(-z));
            ybf[row * D_INNER + d] = f2bf((y + xv * Dv) * (z * sig));
        }
    }
}

// ---------------------------------------------------------------------------
extern "C" void kernel_launch(void* const* d_in, const int* in_sizes, int n_in,
                              void* d_out, int out_size, void* d_ws, size_t ws_size,
                              hipStream_t stream) {
    (void)in_sizes; (void)n_in; (void)out_size; (void)ws_size;
    const float* x        = (const float*)d_in[0];
    const float* fusion_w = (const float*)d_in[19];
    const float* fusion_b = (const float*)d_in[20];

    char* base = (char*)d_ws;
    size_t off = 0;
    auto take = [&](size_t bytes) -> char* {
        char* p = base + off;
        off = (off + bytes + 255) & ~(size_t)255;
        return p;
    };
    float* xz   = (float*)take((size_t)ROWS * 4096 * 4);
    float* xc   = (float*)take((size_t)ROWS * D_INNER * 4);
    float* dbl  = (float*)take((size_t)ROWS * DBL_N * 4);
    float* dl   = (float*)take((size_t)ROWS * D_INNER * 4);         // delta
    unsigned short* xbf    = (unsigned short*)take((size_t)ROWS * D_MODEL * 2);
    unsigned short* xcbf   = (unsigned short*)take((size_t)ROWS * D_INNER * 2);
    unsigned short* dtbf   = (unsigned short*)take((size_t)ROWS * DT_RANK * 2);
    unsigned short* ybf    = (unsigned short*)take((size_t)ROWS * D_INNER * 2);
    unsigned short* combbf = (unsigned short*)take((size_t)ROWS * 2 * D_MODEL * 2);
    unsigned short* wbf_in  = (unsigned short*)take((size_t)4096 * 1024 * 2);
    unsigned short* wbf_x   = (unsigned short*)take((size_t)256 * 2048 * 2);  // 192 padded
    unsigned short* wbf_dt  = (unsigned short*)take((size_t)2048 * 64 * 2);
    unsigned short* wbf_out = (unsigned short*)take((size_t)1024 * 2048 * 2);
    unsigned short* wbf_fus = (unsigned short*)take((size_t)1024 * 2048 * 2);

    // x -> bf16 (GEMM1 A operand), fusion weights -> bf16
    pack_bf16_kernel<<<(ROWS * D_MODEL) / 256, 256, 0, stream>>>(x, xbf);
    pack_weight_kernel<<<(1024 * 2048) / 256, 256, 0, stream>>>(fusion_w, wbf_fus, 1024, 2048);

    for (int dir = 0; dir < 2; ++dir) {
        void* const* p = d_in + 1 + dir * 9;
        const float* in_w   = (const float*)p[0];
        const float* conv_w = (const float*)p[1];
        const float* conv_b = (const float*)p[2];
        const float* x_w    = (const float*)p[3];
        const float* dt_w   = (const float*)p[4];
        const float* dt_b   = (const float*)p[5];
        const float* A_log  = (const float*)p[6];
        const float* Dp     = (const float*)p[7];
        const float* out_w  = (const float*)p[8];

        pack_weight_kernel<<<(4096 * 1024) / 256, 256, 0, stream>>>(in_w, wbf_in, 4096, 1024);
        pack_weight_kernel<<<(256 * 2048) / 256, 256, 0, stream>>>(x_w, wbf_x, 192, 2048);
        pack_weight_kernel<<<(2048 * 64) / 256, 256, 0, stream>>>(dt_w, wbf_dt, 2048, 64);
        pack_weight_kernel<<<(1024 * 2048) / 256, 256, 0, stream>>>(out_w, wbf_out, 1024, 2048);

        // xz = x @ in_w.T                         (8192 x 4096, K=1024)
        gemm_async_wmma<0><<<dim3(32, 64), 256, 0, stream>>>(
            xbf, wbf_in, nullptr, nullptr, xz, 4096, D_MODEL, D_MODEL, 4096);
        // xc = silu(causal_conv(xi) + conv_b)  (f32 + bf16)
        conv_silu_kernel<<<(ROWS * D_INNER) / 256, 256, 0, stream>>>(
            xz, conv_w, conv_b, xc, xcbf, dir);
        // dbl = xc @ x_w.T                        (8192 x 192, K=2048)
        gemm_async_wmma<0><<<dim3(2, 64), 256, 0, stream>>>(
            xcbf, wbf_x, nullptr, nullptr, dbl, DBL_N, D_INNER, D_INNER, DBL_N);
        // dt slice -> bf16
        pack_dt_kernel<<<(ROWS * DT_RANK) / 256, 256, 0, stream>>>(dbl, dtbf);
        // delta = softplus(dt @ dt_w.T + dt_b)    (8192 x 2048, K=64)
        gemm_async_wmma<1><<<dim3(16, 64), 256, 0, stream>>>(
            dtbf, wbf_dt, dt_b, nullptr, dl, D_INNER, DT_RANK, DT_RANK, D_INNER);
        // selective scan + gating epilogue -> ybf (bf16)
        scan_kernel<<<1024, 256, 0, stream>>>(
            dl, dbl, xz, xc, A_log, Dp, ybf, dir);
        // out_dir = y @ out_w.T -> bf16 half of combbf (8192 x 1024, K=2048)
        gemm_async_wmma<3><<<dim3(8, 64), 256, 0, stream>>>(
            ybf, wbf_out, nullptr, nullptr, combbf + dir * D_MODEL,
            D_MODEL, D_INNER, D_INNER, 2 * D_MODEL);
    }
    // out = x + comb @ fusion_w.T + fusion_b      (8192 x 1024, K=2048)
    gemm_async_wmma<2><<<dim3(8, 64), 256, 0, stream>>>(
        combbf, wbf_fus, fusion_b, x, d_out,
        D_MODEL, 2 * D_MODEL, 2 * D_MODEL, D_MODEL);
}
